// BigBirdEncoder_14980845928964
// MI455X (gfx1250) — matrix operand order repaired and checked by reading, not compiled, not run on hardware
//
#include <hip/hip_runtime.h>
#include <hip/hip_bf16.h>

// ---------------------------------------------------------------------------
// BigBird encoder for gfx1250 (CDNA5). All GEMM-shaped math runs through
// v_wmma_f32_16x16x32_bf16 (bf16 operands, f32 accumulate).
// Weights pre-converted f32->bf16 once per launch; GEMM is 128x64x32 tiles
// with double-buffered LDS staging; attention stages V transposed so both
// WMMA operands use paired LDS reads.
// ---------------------------------------------------------------------------

typedef __attribute__((ext_vector_type(16))) __bf16 v16bf;
typedef __attribute__((ext_vector_type(8)))  float  v8f;

#define B_    2
#define L_    4096
#define D_    768
#define H_    12
#define DH_   64
#define MFF_  3072
#define NL_   2
#define BS_   64
#define NB_   (L_ / BS_)      // 64 blocks
#define KB_   8               // gathered key blocks per query block (5 + 3 rand)
#define ROWS_ (B_ * L_)       // 8192
#define NEGV  (-1e9f)

// LDS strides (in halfwords)
#define ASTR_ 40   // GEMM A tile   (row*80B -> 16B aligned, bank-spread)
#define BSTR_ 34   // GEMM B^T tile (even -> 4B-aligned pair reads)
#define QSTR_ 72   // attention q/k/vT tiles (row*144B -> 16B aligned)

static __device__ __forceinline__ unsigned short f2bf(float f) {
  unsigned u = __builtin_bit_cast(unsigned, f);
  u += 0x7fffu + ((u >> 16) & 1u);          // round-to-nearest-even
  return (unsigned short)(u >> 16);
}
static __device__ __forceinline__ __bf16 bfbits(unsigned short u) {
  return __builtin_bit_cast(__bf16, u);
}

static __device__ __forceinline__ v8f wmma_bf16(v16bf a, v16bf b, v8f c) {
  return __builtin_amdgcn_wmma_f32_16x16x32_bf16(false, a, false, b, (short)0, c,
                                                 false, false);
}

// A-operand fragment (16x32 bf16). ISA layout: lane m = lane&15,
// kb = 8*(lane>>4); VGPR pairs hold K = kb + {0..7} then kb + {16..23}.
static __device__ __forceinline__ v16bf frag_a(const unsigned short* rowBase, int kb) {
  v16bf f;
#pragma unroll
  for (int p = 0; p < 8; ++p) {
    const int kOff = kb + ((p < 4) ? (2 * p) : (2 * p + 8));
    const unsigned pr = *(const unsigned*)(rowBase + kOff);
    f[2 * p]     = bfbits((unsigned short)(pr & 0xffffu));
    f[2 * p + 1] = bfbits((unsigned short)(pr >> 16));
  }
  return f;
}
// B-operand fragment (32x16 bf16), LDS holds B^T as [n][k]. ISA layout:
// lane col n = lane&15, kb = 8*(lane>>4); K = kb + {0..7} then kb + {16..23}.
static __device__ __forceinline__ v16bf frag_b(const unsigned short* rowBase, int kb16) {
  v16bf f;
#pragma unroll
  for (int p = 0; p < 8; ++p) {
    const unsigned pr = *(const unsigned*)(rowBase + kb16 + ((p < 4) ? (2 * p) : (2 * p + 8)));
    f[2 * p]     = bfbits((unsigned short)(pr & 0xffffu));
    f[2 * p + 1] = bfbits((unsigned short)(pr >> 16));
  }
  return f;
}

// ---------------------------------------------------------------------------
// Bulk f32 -> bf16 conversion (weights), 8 elems/thread, vectorized.
// ---------------------------------------------------------------------------
__global__ __launch_bounds__(256) void bb_cvt_kernel(
    const float* __restrict__ src, unsigned short* __restrict__ dst) {
  const size_t i = ((size_t)blockIdx.x * 256 + threadIdx.x) * 8;
  const float4 a = *(const float4*)(src + i);
  const float4 b = *(const float4*)(src + i + 4);
  uint4 o;
  o.x = (unsigned)f2bf(a.x) | ((unsigned)f2bf(a.y) << 16);
  o.y = (unsigned)f2bf(a.z) | ((unsigned)f2bf(a.w) << 16);
  o.z = (unsigned)f2bf(b.x) | ((unsigned)f2bf(b.y) << 16);
  o.w = (unsigned)f2bf(b.z) | ((unsigned)f2bf(b.w) << 16);
  *(uint4*)(dst + i) = o;
}

// ---------------------------------------------------------------------------
// Embedding + sinusoidal positional encoding -> x (f32)
// ---------------------------------------------------------------------------
__global__ __launch_bounds__(256) void bb_embed_kernel(
    const int* __restrict__ tokens, const float* __restrict__ emb,
    float* __restrict__ X) {
  const int row = blockIdx.x, t = threadIdx.x;
  const int l = row % L_;
  const int tok = tokens[row];
#pragma unroll
  for (int j = 0; j < 3; ++j) {
    const int d = t + j * 256;
    const int p = (d < D_ / 2) ? d : d - D_ / 2;
    const float dv = __expf((float)(2 * p) * (-9.210340371976184f / (float)D_));
    const float ang = (float)l * dv;
    const float pe = (d < D_ / 2) ? __sinf(ang) : __cosf(ang);
    X[(size_t)row * D_ + d] = emb[(size_t)tok * D_ + d] + pe;
  }
}

// ---------------------------------------------------------------------------
// LayerNorm: one row per workgroup (D=768, 3 elems/thread).
// Emits bf16 (GEMM input) and/or f32 (final output).
// ---------------------------------------------------------------------------
__global__ __launch_bounds__(256) void bb_layernorm_kernel(
    const float* __restrict__ X, const float* __restrict__ g,
    const float* __restrict__ bta, unsigned short* __restrict__ outB,
    float* __restrict__ outF) {
  __shared__ float red[8];
  __shared__ float bc[2];
  const int row = blockIdx.x, t = threadIdx.x;
  const int wid = t >> 5, lane = t & 31;
  const float* xr = X + (size_t)row * D_;
  const float v0 = xr[t], v1 = xr[t + 256], v2 = xr[t + 512];
  float s = v0 + v1 + v2;
#pragma unroll
  for (int d = 16; d >= 1; d >>= 1) s += __shfl_xor(s, d, 32);
  if (lane == 0) red[wid] = s;
  __syncthreads();
  if (t == 0) {
    float tot = 0.f;
#pragma unroll
    for (int j = 0; j < 8; ++j) tot += red[j];
    bc[0] = tot * (1.0f / (float)D_);
  }
  __syncthreads();
  const float mean = bc[0];
  const float d0 = v0 - mean, d1 = v1 - mean, d2 = v2 - mean;
  float q = d0 * d0 + d1 * d1 + d2 * d2;
#pragma unroll
  for (int d = 16; d >= 1; d >>= 1) q += __shfl_xor(q, d, 32);
  if (lane == 0) red[wid] = q;
  __syncthreads();
  if (t == 0) {
    float tot = 0.f;
#pragma unroll
    for (int j = 0; j < 8; ++j) tot += red[j];
    bc[1] = tot * (1.0f / (float)D_);
  }
  __syncthreads();
  const float inv = rsqrtf(bc[1] + 1e-6f);
#pragma unroll
  for (int j = 0; j < 3; ++j) {
    const int c = t + j * 256;
    const float dd = (j == 0) ? d0 : (j == 1) ? d1 : d2;
    const float y = dd * inv * g[c] + bta[c];
    if (outB) outB[(size_t)row * D_ + c] = f2bf(y);
    if (outF) outF[(size_t)row * D_ + c] = y;
  }
}

// ---------------------------------------------------------------------------
// Stage one (A 128x32, B^T 64x32) tile pair into LDS. A and B are bf16.
// ---------------------------------------------------------------------------
static __device__ __forceinline__ void stage_tiles(
    unsigned short* ldsA, unsigned short* ldsB,
    const unsigned short* __restrict__ A, const unsigned short* __restrict__ Bw,
    int m0, int n0, int k0, int N, int K, int t) {
  // A: 128 rows x 32 cols; 2 threads/row, 16 contiguous bf16 each (2x b128)
  {
    const int arow = t >> 1, acol = (t & 1) * 16;
    const uint4* sa = (const uint4*)(A + (size_t)(m0 + arow) * K + (k0 + acol));
    uint4* da = (uint4*)(ldsA + arow * ASTR_ + acol);
    da[0] = sa[0];
    da[1] = sa[1];
  }
  // B: 32 rows x 64 cols, transpose into [n][k]
  {
    const int bk = t >> 3, bn = (t & 7) * 8;
    const uint4 bv = *(const uint4*)(Bw + (size_t)(k0 + bk) * N + (n0 + bn));
    const unsigned w[4] = {bv.x, bv.y, bv.z, bv.w};
    unsigned short* db = ldsB + bk;
#pragma unroll
    for (int j = 0; j < 4; ++j) {
      db[(bn + 2 * j) * BSTR_]     = (unsigned short)(w[j] & 0xffffu);
      db[(bn + 2 * j + 1) * BSTR_] = (unsigned short)(w[j] >> 16);
    }
  }
}

// ---------------------------------------------------------------------------
// WMMA GEMM:  C[M,N] = alpha * A(bf16,[M,K]) * Bw(bf16,[K,N]) (+bias)
//             (+ReLU) (+resid f32)  -> outF (f32) and/or outB (bf16)
// Tile 128x64, BK=32, double-buffered LDS; 8 waves (2x4), 4 accum/wave.
// All fragment loads are hoisted ahead of the WMMA burst so the DS waits
// amortize across 4 back-to-back v_wmma.
// ---------------------------------------------------------------------------
__global__ __launch_bounds__(256) void bb_gemm_kernel(
    const unsigned short* __restrict__ A, const unsigned short* __restrict__ Bw,
    const float* __restrict__ bias, const float* __restrict__ resid,
    float* __restrict__ outF, unsigned short* __restrict__ outB,
    int N, int K, float alpha, int relu) {
  __shared__ __align__(16) unsigned short ldsA[2][128 * ASTR_];
  __shared__ __align__(16) unsigned short ldsB[2][64 * BSTR_];
  const int t = threadIdx.x;
  const int wid = t >> 5, lane = t & 31, lh = lane >> 4, ln = lane & 15;
  const int wm = wid >> 2, wn = wid & 3;       // wm: 0..1 (64 rows each)
  const int m0 = blockIdx.y * 128, n0 = blockIdx.x * 64;

  v8f acc[4] = {{}, {}, {}, {}};
  stage_tiles(ldsA[0], ldsB[0], A, Bw, m0, n0, 0, N, K, t);
  __syncthreads();
  int buf = 0;
  for (int k0 = 0; k0 < K; k0 += 32) {
    if (k0 + 32 < K)   // prefetch next tile into the other buffer
      stage_tiles(ldsA[buf ^ 1], ldsB[buf ^ 1], A, Bw, m0, n0, k0 + 32, N, K, t);
    // gather all fragments first, then issue the 4 WMMAs back-to-back
    const v16bf bf = frag_b(ldsB[buf] + (wn * 16 + ln) * BSTR_, lh * 8);
    v16bf afr[4];
#pragma unroll
    for (int s = 0; s < 4; ++s)
      afr[s] = frag_a(ldsA[buf] + (wm * 64 + s * 16 + ln) * ASTR_, lh * 8);
#pragma unroll
    for (int s = 0; s < 4; ++s)
      acc[s] = wmma_bf16(afr[s], bf, acc[s]);
    __syncthreads();
    buf ^= 1;
  }

  const int ncol = n0 + wn * 16 + ln;
  const float bv = bias ? bias[ncol] : 0.0f;
#pragma unroll
  for (int s = 0; s < 4; ++s) {
    const int mb = m0 + wm * 64 + s * 16 + lh * 8;
#pragma unroll
    for (int r = 0; r < 8; ++r) {
      float v = acc[s][r] * alpha + bv;
      if (relu) v = fmaxf(v, 0.0f);
      const size_t off = (size_t)(mb + r) * N + ncol;
      if (resid) v += resid[off];
      if (outF) outF[off] = v;
      if (outB) outB[off] = f2bf(v);
    }
  }
}

// ---------------------------------------------------------------------------
// Flash-style attention. mode 0: block-sparse (8 gathered key blocks per
// query block, skips q-blocks 0 and NB-1). mode 1: global rows (q-blocks 0
// and NB-1 attend over all NB key blocks, pad mask only).
// Workgroup = (b, h, q-block); 128 threads = 4 waves x 16 query rows.
// V is staged TRANSPOSED (ldsVT[dh][key]) so the P*V B-operand uses the
// same paired-read frag_b path as Q*K^T.
// ---------------------------------------------------------------------------
__global__ __launch_bounds__(128) void bb_attn_kernel(
    const unsigned short* __restrict__ Qp, const unsigned short* __restrict__ Kp,
    const unsigned short* __restrict__ Vp, const int* __restrict__ tokens,
    const int* __restrict__ bidx, const unsigned char* __restrict__ bval,
    unsigned short* __restrict__ Optr, int layer, int mode) {
  __shared__ __align__(16) unsigned short ldsQ[64 * QSTR_];
  __shared__ __align__(16) unsigned short ldsK[64 * QSTR_];
  __shared__ __align__(16) unsigned short ldsVT[64 * QSTR_];  // [dh][key]
  __shared__ __align__(16) unsigned short ldsP[4][16 * QSTR_];
  __shared__ unsigned char kmask[64];

  int i;
  if (mode == 0) {
    i = blockIdx.x;
    if (i == 0 || i == NB_ - 1) return;  // overwritten by global pass
  } else {
    i = blockIdx.x ? (NB_ - 1) : 0;
  }
  const int h = blockIdx.y, b = blockIdx.z;
  const int nblk = mode ? NB_ : KB_;
  const int t = threadIdx.x;
  const int wid = t >> 5, lane = t & 31, lh = lane >> 4, ln = lane & 15;
  const int srow = t >> 1, shalf = t & 1;   // staging: 2 threads/row, 32 elems

  // stage q block (bf16 rows, head slice)
  {
    const uint4* s = (const uint4*)(Qp + ((size_t)(b * L_ + i * BS_ + srow)) * D_ +
                                    h * DH_ + shalf * 32);
    uint4* d = (uint4*)(ldsQ + srow * QSTR_ + shalf * 32);
    d[0] = s[0]; d[1] = s[1]; d[2] = s[2]; d[3] = s[3];
  }
  __syncthreads();
  const v16bf aq0 = frag_a(ldsQ + (wid * 16 + ln) * QSTR_, lh * 8);
  const v16bf aq1 = frag_a(ldsQ + (wid * 16 + ln) * QSTR_ + 32, lh * 8);

  v8f o0 = {}, o1 = {}, o2 = {}, o3 = {};
  float mrun[8], lrun[8];
#pragma unroll
  for (int r = 0; r < 8; ++r) { mrun[r] = -3.0e38f; lrun[r] = 0.0f; }

  for (int jj = 0; jj < nblk; ++jj) {
    int blk, vld;
    if (mode == 0) {
      const int idx0 = (layer * NB_ + i) * KB_ + jj;
      blk = bidx[idx0];
      vld = (int)bval[idx0];
    } else {
      blk = jj; vld = 1;
    }
    __syncthreads();  // previous iteration done with ldsK/ldsVT
    {
      const size_t base =
          ((size_t)(b * L_ + blk * BS_ + srow)) * D_ + h * DH_ + shalf * 32;
      // K block: natural [keyrow][dh] layout (vectorized)
      const uint4* sk = (const uint4*)(Kp + base);
      uint4* dk = (uint4*)(ldsK + srow * QSTR_ + shalf * 32);
      dk[0] = sk[0]; dk[1] = sk[1]; dk[2] = sk[2]; dk[3] = sk[3];
      // V block: transpose into [dh][keyrow] (scalar scatter stores)
      const uint4 vv[4] = {((const uint4*)(Vp + base))[0],
                           ((const uint4*)(Vp + base))[1],
                           ((const uint4*)(Vp + base))[2],
                           ((const uint4*)(Vp + base))[3]};
#pragma unroll
      for (int w = 0; w < 4; ++w) {
        const unsigned ww[4] = {vv[w].x, vv[w].y, vv[w].z, vv[w].w};
#pragma unroll
        for (int c = 0; c < 4; ++c) {
          const int dd = shalf * 32 + w * 8 + c * 2;
          ldsVT[dd * QSTR_ + srow]       = (unsigned short)(ww[c] & 0xffffu);
          ldsVT[(dd + 1) * QSTR_ + srow] = (unsigned short)(ww[c] >> 16);
        }
      }
    }
    if (t < 64)
      kmask[t] = (unsigned char)(vld && (tokens[(size_t)b * L_ + blk * BS_ + t] > 0));
    __syncthreads();

    // S = q * k^T  (k block natural [keyrow][dh] layout == B^T layout)
    float sv4[4][8];
#pragma unroll
    for (int nt = 0; nt < 4; ++nt) {
      const v16bf bk0 = frag_b(ldsK + (nt * 16 + ln) * QSTR_, lh * 8);
      const v16bf bk1 = frag_b(ldsK + (nt * 16 + ln) * QSTR_ + 32, lh * 8);
      v8f s = {};
      s = wmma_bf16(aq0, bk0, s);
      s = wmma_bf16(aq1, bk1, s);
      const bool ok = kmask[nt * 16 + ln] != 0;
#pragma unroll
      for (int r = 0; r < 8; ++r) sv4[nt][r] = ok ? s[r] : NEGV;
    }

    // online softmax over this 64-key block
#pragma unroll
    for (int r = 0; r < 8; ++r) {
      float mx = fmaxf(fmaxf(sv4[0][r], sv4[1][r]), fmaxf(sv4[2][r], sv4[3][r]));
#pragma unroll
      for (int dlt = 8; dlt >= 1; dlt >>= 1) mx = fmaxf(mx, __shfl_xor(mx, dlt, 32));
      const float mnew = fmaxf(mrun[r], mx);
      const float sc = __expf(mrun[r] - mnew);
      float rs = 0.0f;
#pragma unroll
      for (int nt = 0; nt < 4; ++nt) {
        const float p = __expf(sv4[nt][r] - mnew);
        sv4[nt][r] = p;
        rs += p;
      }
#pragma unroll
      for (int dlt = 8; dlt >= 1; dlt >>= 1) rs += __shfl_xor(rs, dlt, 32);
      lrun[r] = lrun[r] * sc + rs;
      mrun[r] = mnew;
      o0[r] *= sc; o1[r] *= sc; o2[r] *= sc; o3[r] *= sc;
    }

    // spill P (C layout) to LDS, re-gather as A fragment
#pragma unroll
    for (int nt = 0; nt < 4; ++nt)
#pragma unroll
      for (int r = 0; r < 8; ++r)
        ldsP[wid][(r + lh * 8) * QSTR_ + nt * 16 + ln] = f2bf(sv4[nt][r]);
    __syncthreads();

    // O += P * V   (B-operand from transposed V: ldsVT[n=dh][k=key])
#pragma unroll
    for (int c = 0; c < 2; ++c) {
      const v16bf ap = frag_a(ldsP[wid] + ln * QSTR_ + c * 32, lh * 8);
#pragma unroll
      for (int nt = 0; nt < 4; ++nt) {
        const v16bf bvv =
            frag_b(ldsVT + (nt * 16 + ln) * QSTR_ + c * 32, lh * 8);
        if (nt == 0)      o0 = wmma_bf16(ap, bvv, o0);
        else if (nt == 1) o1 = wmma_bf16(ap, bvv, o1);
        else if (nt == 2) o2 = wmma_bf16(ap, bvv, o2);
        else              o3 = wmma_bf16(ap, bvv, o3);
      }
    }
  }

  float inv[8];
#pragma unroll
  for (int r = 0; r < 8; ++r) inv[r] = (lrun[r] > 0.0f) ? 1.0f / lrun[r] : 0.0f;
#pragma unroll
  for (int nt = 0; nt < 4; ++nt) {
    const v8f oo = (nt == 0) ? o0 : (nt == 1) ? o1 : (nt == 2) ? o2 : o3;
#pragma unroll
    for (int r = 0; r < 8; ++r) {
      const int row = i * BS_ + wid * 16 + r + lh * 8;
      const int col = h * DH_ + nt * 16 + ln;
      Optr[(size_t)(b * L_ + row) * D_ + col] = f2bf(oo[r] * inv[r]);
    }
  }
}

// ---------------------------------------------------------------------------
// Host-side launch sequence (graph-capture safe; stream only)
// ---------------------------------------------------------------------------
extern "C" void kernel_launch(void* const* d_in, const int* in_sizes, int n_in,
                              void* d_out, int out_size, void* d_ws, size_t ws_size,
                              hipStream_t stream) {
  (void)in_sizes; (void)n_in; (void)out_size; (void)ws_size;
  const int*   tokens = (const int*)  d_in[0];
  const float* emb    = (const float*)d_in[1];
  const float* ln1s   = (const float*)d_in[2];
  const float* ln1b   = (const float*)d_in[3];
  const float* wq     = (const float*)d_in[4];
  const float* wk     = (const float*)d_in[5];
  const float* wv     = (const float*)d_in[6];
  const float* wo     = (const float*)d_in[7];
  const float* ln2s   = (const float*)d_in[8];
  const float* ln2b   = (const float*)d_in[9];
  const float* w1     = (const float*)d_in[10];
  const float* b1     = (const float*)d_in[11];
  const float* w2     = (const float*)d_in[12];
  const float* b2     = (const float*)d_in[13];
  const float* lnfs   = (const float*)d_in[14];
  const float* lnfb   = (const float*)d_in[15];
  const int*   bidx   = (const int*)  d_in[16];
  const unsigned char* bvalid = (const unsigned char*)d_in[17];  // jnp bool_

  char* ws = (char*)d_ws;
  size_t off = 0;
  auto alloc = [&](size_t bytes) -> char* {
    char* p = ws + off;
    off += (bytes + 255) & ~(size_t)255;
    return p;
  };
  float*          x      = (float*)alloc((size_t)ROWS_ * D_ * sizeof(float));
  unsigned short* ybf    = (unsigned short*)alloc((size_t)ROWS_ * D_ * 2);
  unsigned short* qbf    = (unsigned short*)alloc((size_t)ROWS_ * D_ * 2);
  unsigned short* kbf    = (unsigned short*)alloc((size_t)ROWS_ * D_ * 2);
  unsigned short* vbf    = (unsigned short*)alloc((size_t)ROWS_ * D_ * 2);
  unsigned short* attnbf = (unsigned short*)alloc((size_t)ROWS_ * D_ * 2);
  unsigned short* hbf    = (unsigned short*)alloc((size_t)ROWS_ * MFF_ * 2);
  // bf16 weight copies (converted once per launch)
  const size_t SQ = (size_t)NL_ * D_ * D_;     // wq/wk/wv/wo each
  const size_t SF = (size_t)NL_ * D_ * MFF_;   // w1/w2 each
  unsigned short* wqb = (unsigned short*)alloc(SQ * 2);
  unsigned short* wkb = (unsigned short*)alloc(SQ * 2);
  unsigned short* wvb = (unsigned short*)alloc(SQ * 2);
  unsigned short* wob = (unsigned short*)alloc(SQ * 2);
  unsigned short* w1b = (unsigned short*)alloc(SF * 2);
  unsigned short* w2b = (unsigned short*)alloc(SF * 2);

  // weight conversion pre-pass (n divisible by 2048 for all tensors)
  bb_cvt_kernel<<<(unsigned)(SQ / 2048), 256, 0, stream>>>(wq, wqb);
  bb_cvt_kernel<<<(unsigned)(SQ / 2048), 256, 0, stream>>>(wk, wkb);
  bb_cvt_kernel<<<(unsigned)(SQ / 2048), 256, 0, stream>>>(wv, wvb);
  bb_cvt_kernel<<<(unsigned)(SQ / 2048), 256, 0, stream>>>(wo, wob);
  bb_cvt_kernel<<<(unsigned)(SF / 2048), 256, 0, stream>>>(w1, w1b);
  bb_cvt_kernel<<<(unsigned)(SF / 2048), 256, 0, stream>>>(w2, w2b);

  const dim3 g768(D_ / 64, ROWS_ / 128);
  const dim3 gff(MFF_ / 64, ROWS_ / 128);

  bb_embed_kernel<<<ROWS_, 256, 0, stream>>>(tokens, emb, x);

  for (int l = 0; l < NL_; ++l) {
    bb_layernorm_kernel<<<ROWS_, 256, 0, stream>>>(x, ln1s + l * D_, ln1b + l * D_,
                                                   ybf, nullptr);
    // Q (scaled by dh^-0.5 = 0.125), K, V projections -> bf16
    bb_gemm_kernel<<<g768, 256, 0, stream>>>(ybf, wqb + (size_t)l * D_ * D_, nullptr,
                                             nullptr, nullptr, qbf, D_, D_, 0.125f, 0);
    bb_gemm_kernel<<<g768, 256, 0, stream>>>(ybf, wkb + (size_t)l * D_ * D_, nullptr,
                                             nullptr, nullptr, kbf, D_, D_, 1.0f, 0);
    bb_gemm_kernel<<<g768, 256, 0, stream>>>(ybf, wvb + (size_t)l * D_ * D_, nullptr,
                                             nullptr, nullptr, vbf, D_, D_, 1.0f, 0);
    // block-sparse pass, then global rows overwrite q-blocks 0 and NB-1
    bb_attn_kernel<<<dim3(NB_, H_, B_), 128, 0, stream>>>(qbf, kbf, vbf, tokens,
                                                          bidx, bvalid, attnbf, l, 0);
    bb_attn_kernel<<<dim3(2, H_, B_), 128, 0, stream>>>(qbf, kbf, vbf, tokens,
                                                        bidx, bvalid, attnbf, l, 1);
    // output projection + residual -> x (f32)
    bb_gemm_kernel<<<g768, 256, 0, stream>>>(attnbf, wob + (size_t)l * D_ * D_,
                                             nullptr, x, x, nullptr, D_, D_, 1.0f, 0);
    // FFN
    bb_layernorm_kernel<<<ROWS_, 256, 0, stream>>>(x, ln2s + l * D_, ln2b + l * D_,
                                                   ybf, nullptr);
    bb_gemm_kernel<<<gff, 256, 0, stream>>>(ybf, w1b + (size_t)l * D_ * MFF_,
                                            b1 + (size_t)l * MFF_, nullptr, nullptr,
                                            hbf, MFF_, D_, 1.0f, 1);
    bb_gemm_kernel<<<g768, 256, 0, stream>>>(hbf, w2b + (size_t)l * MFF_ * D_,
                                             b2 + (size_t)l * D_, x, x, nullptr,
                                             D_, MFF_, 1.0f, 0);
  }
  bb_layernorm_kernel<<<ROWS_, 256, 0, stream>>>(x, lnfs, lnfb, nullptr,
                                                 (float*)d_out);
}
